// NAVAR_31353261260861
// MI455X (gfx1250) — matrix operand order repaired and checked by reading, not compiled, not run on hardware
//
#include <hip/hip_runtime.h>

typedef __attribute__((ext_vector_type(16))) _Float16 v16h;
typedef __attribute__((ext_vector_type(8)))  float    v8f;

#define ND_     2
#define NS_     4
#define NV_     12
#define HID_    16
#define GROUPS_ 96
#define KW      4
#define NBLK    3
#define BATCH_  8
#define TT      2048
#define TILE_   256
#define BOFF    48                        // buffer time offset (halo + guard)
#define HBT     (TILE_ + BOFF)            // 304 time slots per LDS buffer
#define PRED_SIZE (BATCH_*ND_*NS_*NV_*TT) // 1,572,864 floats

union V16 { v16h v; _Float16 e[16]; };
union V8F { v8f v; float f[8]; };
union H8  { _Float16 h[8]; uint4 u; };

// One workgroup = one (batch, group, 256-time tile). 4 waves cooperate
// layer-by-layer through LDS ping-pong buffers; hidden 16->16 channel
// mixes and the 1x1 output conv run on v_wmma_f32_16x16x32_f16.
__global__ __launch_bounds__(128)
void navar_main(const float* __restrict__ x,
                const float* __restrict__ w_in, const float* __restrict__ b_in,
                const float* __restrict__ w_h,  const float* __restrict__ b_h,
                const float* __restrict__ w_out,const float* __restrict__ b_out,
                float* __restrict__ contrib)
{
    __shared__ __align__(32) _Float16 hbuf[2][HBT * HID_]; // [t][c] f16 rows (32B)
    __shared__ float    xs[HBT];
    __shared__ _Float16 wblk[HID_ * HID_ * KW];

    const int tid  = threadIdx.x;
    const int lane = tid & 31;
    // scalar (SGPR) wave index -> uniform loops, EXEC stays all-1s around WMMA
    const int wave = __builtin_amdgcn_readfirstlane(tid >> 5);
    const int j    = lane & 15;   // A: M row / B: N (time) column / D: N column
    const int half = lane >> 4;   // lane-half select per WMMA layouts

    const int ntile = blockIdx.x % (TT / TILE_);
    const int strm  = blockIdx.x / (TT / TILE_);
    const int g     = strm % GROUPS_;
    const int b     = strm / GROUPS_;
    const int s0    = ntile * TILE_;

    // ---- stage x tile (causal zero padding for t<0) ----
    const float* xg = x + (size_t)(b * GROUPS_ + g) * TT;
    for (int bt = tid; bt < HBT; bt += 128) {
        int t = s0 - BOFF + bt;
        xs[bt] = (t >= 0) ? xg[t] : 0.0f;
    }
    // zero the low guard region of both h buffers (read-only halo below bt=16)
    for (int i = tid; i < 2 * 16 * HID_; i += 128)
        hbuf[i / (16 * HID_)][i % (16 * HID_)] = (_Float16)0.0f;
    __syncthreads();

    // ---- input conv (1 -> HID, K=4, d=1): tiny, VALU fp32 ----
    {
        const float* wi = w_in + (size_t)g * HID_ * KW;
        const float* bi = b_in + (size_t)g * HID_;
        for (int idx = tid; idx < (HBT - 16) * HID_; idx += 128) {
            int c  = idx & (HID_ - 1);
            int bt = 16 + (idx >> 4);
            float acc = bi[c];
            #pragma unroll
            for (int k = 0; k < KW; ++k)
                acc = fmaf(wi[c * KW + k], xs[bt - k], acc);
            int t = s0 - BOFF + bt;
            hbuf[0][bt * HID_ + c] = (_Float16)((t < 0) ? 0.0f : fmaxf(acc, 0.0f));
        }
    }
    __syncthreads();

    // ---- 3 dilated blocks (HID->HID, K=4, d=1,2,4) on WMMA ----
    for (int blk = 0; blk < NBLK; ++blk) {
        const int d    = 1 << blk;
        const int srcb = blk & 1;     // 0,1,0
        const int dstb = srcb ^ 1;    // 1,0,1

        const float* wh = w_h + (size_t)(blk * GROUPS_ + g) * HID_ * HID_ * KW;
        for (int i = tid; i < HID_ * HID_ * KW; i += 128)
            wblk[i] = (_Float16)wh[i];
        __syncthreads();

        // A fragments: M=out channel (=j), K64=(tap,chan); chunk0=taps0-1, chunk1=taps2-3.
        // A 16x32 layout (contiguous, matches 32B B rows): K = 16*half + i.
        V16 A0, A1;
        #pragma unroll
        for (int i = 0; i < 16; ++i) {
            int Kidx = half * 16 + i;
            int c   = Kidx & 15;
            int khi = Kidx >> 4;                       // tap within chunk
            A0.e[i] = wblk[(j * HID_ + c) * KW + khi];
            A1.e[i] = wblk[(j * HID_ + c) * KW + 2 + khi];
        }

        // per-lane biases (channels M = 8*half + r), hoisted out of tile loop
        const float* bh = b_h + (size_t)blk * GROUPS_ * HID_ + (size_t)g * HID_;
        float biasr[8];
        #pragma unroll
        for (int r = 0; r < 8; ++r) biasr[r] = bh[8 * half + r];

        const int kB0 = half;        // chunk0 tap index = 2*0 + half
        const int kB1 = 2 + half;    // chunk1 tap index = 2*1 + half

        for (int tile = wave; tile < (HBT - 16) / 16; tile += 4) {  // 18 tiles, scalar loop
            int t0 = 16 + 16 * tile;
            // issue both B fragments before the WMMA pair -> single dscnt wait
            V16 B0, B1;
            B0.v = *(const v16h*)&hbuf[srcb][(t0 + j - kB0 * d) * HID_];
            B1.v = *(const v16h*)&hbuf[srcb][(t0 + j - kB1 * d) * HID_];
            v8f acc = {};
            acc = __builtin_amdgcn_wmma_f32_16x16x32_f16(false, A0.v, false, B0.v,
                                                         (short)0, acc, false, false);
            acc = __builtin_amdgcn_wmma_f32_16x16x32_f16(false, A1.v, false, B1.v,
                                                         (short)0, acc, false, false);
            V8F C; C.v = acc;
            int t = s0 - BOFF + t0 + j;
            H8 o;
            #pragma unroll
            for (int r = 0; r < 8; ++r) {              // D: M = 8*half + r
                float y = fmaxf(C.f[r] + biasr[r], 0.0f);
                o.h[r] = (_Float16)((t < 0) ? 0.0f : y);
            }
            *(uint4*)&hbuf[dstb][(t0 + j) * HID_ + 8 * half] = o.u;  // 16B store
        }
        __syncthreads();
    }

    // ---- 1x1 output conv (HID -> NV) on WMMA, K padded 16->32, M padded 12->16 ----
    {
        // A zero-padded for K>=16 and M>=12; B rows K>=16 may hold any finite
        // data (they multiply the zero columns of A), so all lanes load the
        // same 32B activation row -> no divergent branch near the WMMA.
        V16 Ao;
        #pragma unroll
        for (int i = 0; i < 16; ++i) {
            int Kidx = half * 16 + i;
            float wv = 0.0f;
            if (Kidx < HID_ && j < NV_)
                wv = w_out[((size_t)g * NV_ + j) * HID_ + Kidx];
            Ao.e[i] = (_Float16)wv;
        }
        const int samp = b * (ND_ * NS_) + g / NV_;
        const int srcv = g % NV_;
        float* cb = contrib + (size_t)samp * NV_ * NV_ * TT;
        const float* bo = b_out + (size_t)g * NV_;
        float bor[8];
        #pragma unroll
        for (int r = 0; r < 8; ++r) {
            int dc = 8 * half + r;
            bor[r] = (dc < NV_) ? bo[dc] : 0.0f;
        }

        for (int tile = wave; tile < TILE_ / 16; tile += 4) {  // 16 tiles, scalar loop
            int t0 = BOFF + 16 * tile;
            V16 Bo;
            Bo.v = *(const v16h*)&hbuf[1][(t0 + j) * HID_];
            v8f acc = {};
            acc = __builtin_amdgcn_wmma_f32_16x16x32_f16(false, Ao.v, false, Bo.v,
                                                         (short)0, acc, false, false);
            V8F C; C.v = acc;
            int t = s0 + (t0 - BOFF) + j;
            #pragma unroll
            for (int r = 0; r < 8; ++r) {
                int dc = 8 * half + r;                 // dst channel
                if (dc < NV_)
                    cb[((size_t)dc * NV_ + srcv) * TT + t] = C.f[r] + bor[r];
            }
        }
    }
}

// prediction[b,nd,ns,dst,t] = sum_src contributions[b,nd,ns,dst,src,t] + bias
// (contributions were just written: 75 MB fits in the 192 MB L2, so this
// pass reads from L2, not HBM)
__global__ __launch_bounds__(256)
void navar_pred(const float* __restrict__ contrib, const float* __restrict__ biases,
                float* __restrict__ pred)
{
    int i = blockIdx.x * blockDim.x + threadIdx.x;
    if (i >= PRED_SIZE) return;
    int t    = i % TT;
    int rest = i / TT;
    int dstc = rest % NV_;
    int samp = rest / NV_;                 // = b*8 + (nd*NS+ns)
    const float* cb = contrib + ((size_t)samp * NV_ + dstc) * NV_ * TT + t;
    float acc = biases[(samp & 7) * NV_ + dstc];
    #pragma unroll
    for (int sv = 0; sv < NV_; ++sv)
        acc += cb[(size_t)sv * TT];
    pred[i] = acc;
}

extern "C" void kernel_launch(void* const* d_in, const int* in_sizes, int n_in,
                              void* d_out, int out_size, void* d_ws, size_t ws_size,
                              hipStream_t stream)
{
    (void)in_sizes; (void)n_in; (void)out_size; (void)d_ws; (void)ws_size;
    const float* x     = (const float*)d_in[0];
    const float* w_in  = (const float*)d_in[1];
    const float* b_in  = (const float*)d_in[2];
    const float* w_h   = (const float*)d_in[3];
    const float* b_h   = (const float*)d_in[4];
    const float* w_out = (const float*)d_in[5];
    const float* b_out = (const float*)d_in[6];
    const float* bias  = (const float*)d_in[7];

    float* pred    = (float*)d_out;
    float* contrib = pred + PRED_SIZE;     // outputs concatenated: pred, contributions

    dim3 g1(BATCH_ * GROUPS_ * (TT / TILE_));   // 6144 workgroups
    navar_main<<<g1, 128, 0, stream>>>(x, w_in, b_in, w_h, b_h, w_out, b_out, contrib);

    dim3 g2((PRED_SIZE + 255) / 256);
    navar_pred<<<g2, 256, 0, stream>>>(contrib, bias, pred);
}
